// Diffeomorpism_44229573214415
// MI455X (gfx1250) — matrix operand order repaired and checked by reading, not compiled, not run on hardware
//
#include <hip/hip_runtime.h>

typedef _Float16 v16h __attribute__((ext_vector_type(16)));
typedef _Float16 v8h  __attribute__((ext_vector_type(8)));
typedef float    v8f  __attribute__((ext_vector_type(8)));
typedef unsigned int u32x4 __attribute__((ext_vector_type(4)));
typedef int          i32x8 __attribute__((ext_vector_type(8)));
typedef int          i32x4 __attribute__((ext_vector_type(4)));

#define S_DIM 256
#define H_DIM 32
#define L_LAYERS 3
#define XSTR 513            // x / out row stride (D+1)
#define WROWS 128           // rows per workgroup (8 waves x 16)
#define AST 264             // activation row stride in halves (LDS bank pad)
#define W1ST 264            // W1^T row stride (k index, 256 + pad)
#define W2ST 136            // W2^T row stride (hcat index, 128 + pad)
#define TST  72             // tanh staging row stride (64 + pad)

// per-layer weight blob (byte-exact LDS image): w1t | w2t | b1s | b2s
#define W1T_BYTES  (128 * W1ST * 2)              // 67584
#define W2T_BYTES  (256 * W2ST * 2)              // 69632
#define B1_BYTES   (128 * 4)                     // 512
#define B2_BYTES   (1024 * 4)                    // 4096
#define BLOB_BYTES (W1T_BYTES + W2T_BYTES + B1_BYTES + B2_BYTES)   // 141824 (8B aligned)
#define BLOB_QW    (BLOB_BYTES / 8)              // 17728 (< 65536 -> one TDM tile)

// LDS map (bytes)
#define LDS_ACT1   0
#define LDS_ACT2   (WROWS * AST * 2)             // 67584
#define LDS_WBLOB  (2 * WROWS * AST * 2)         // 135168
#define LDS_TSTG   (LDS_WBLOB + BLOB_BYTES)      // 276992
#define LDS_TOTAL  (LDS_TSTG + 8 * 16 * TST * 2) // 295424 (<= 320KB/WGP)

__device__ __forceinline__ float fast_exp(float x) {
    return __builtin_amdgcn_exp2f(x * 1.4426950408889634f);
}
__device__ __forceinline__ float fast_tanh(float x) {
    float e = __builtin_amdgcn_exp2f(x * 2.8853900817779268f); // exp(2x)
    return 1.0f - 2.0f * __builtin_amdgcn_rcpf(e + 1.0f);
}

// ---------------------------------------------------------------------------
// Prep: build per-layer f16 weight blobs in d_ws (exact LDS images, TDM-ready).
// ---------------------------------------------------------------------------
__global__ void prep_weights(const float* __restrict__ W1, const float* __restrict__ b1,
                             const float* __restrict__ W2, const float* __restrict__ b2,
                             unsigned char* __restrict__ ws) {
    const int j = blockIdx.x;                    // layer
    unsigned char* blob = ws + (size_t)j * BLOB_BYTES;
    _Float16* w1t = (_Float16*)blob;
    _Float16* w2t = (_Float16*)(blob + W1T_BYTES);
    float*    b1s = (float*)(blob + W1T_BYTES + W2T_BYTES);
    float*    b2s = b1s + 128;

    const float* w1g = W1 + (size_t)j * 4 * S_DIM * H_DIM;   // [4][256][32]
    for (int i = threadIdx.x; i < 4 * S_DIM * H_DIM; i += blockDim.x) {
        const int mlp = i >> 13, k = (i >> 5) & 255, h = i & 31;
        w1t[(mlp * 32 + h) * W1ST + k] = (_Float16)w1g[i];
    }
    const float* w2g = W2 + (size_t)j * 4 * H_DIM * S_DIM;   // [4][32][256]
    for (int i = threadIdx.x; i < 4 * H_DIM * S_DIM; i += blockDim.x) {
        const int mlp = i >> 13, h = (i >> 8) & 31, s = i & 255;
        w2t[s * W2ST + mlp * 32 + h] = (_Float16)w2g[i];
    }
    for (int i = threadIdx.x; i < 128;  i += blockDim.x) b1s[i] = b1[j * 128 + i];
    for (int i = threadIdx.x; i < 1024; i += blockDim.x) b2s[i] = b2[j * 1024 + i];
}

// ---------------------------------------------------------------------------
// TDM: DMA one per-layer blob (global, f16-prepped) -> LDS weight region.
// D# per CDNA5 ISA (2-D tile, groups 2/3 zero).
// 6-arg builtin form (clang-23 / therock-10.0 headers).
// ---------------------------------------------------------------------------
__device__ __forceinline__ void tdm_load_blob(const unsigned char* gsrc, unsigned lds_off) {
    const unsigned long long ga = (unsigned long long)(uintptr_t)gsrc;
    u32x4 g0;
    g0[0] = 1u;                                          // count=1, is_restore=0, gather=0
    g0[1] = lds_off;                                     // lds_addr (bytes)
    g0[2] = (unsigned)(ga & 0xffffffffull);              // global_addr[31:0]
    g0[3] = (unsigned)((ga >> 32) & 0x01ffffffull)       // global_addr[56:32]
          | 0x80000000u;                                 // type=2 ("image")
    i32x8 g1;
    g1[0] = 0x00030000;                                  // data_size=3 (8B), mask=0, no flags
    g1[1] = (int)((BLOB_QW & 0xffff) << 16);             // tensor_dim0[15:0] (barrier addr 0)
    g1[2] = (int)((BLOB_QW >> 16) | (1u << 16));         // tensor_dim0[31:16] | tensor_dim1[15:0]=1
    g1[3] = (int)((BLOB_QW & 0xffff) << 16);             // tensor_dim1[31:16]=0 | tile_dim0
    g1[4] = 1;                                           // tile_dim1=1 | tile_dim2=0
    g1[5] = (int)BLOB_QW;                                // tensor_dim0_stride[31:0]
    g1[6] = 0;                                           // stride0[47:32] | stride1[15:0]
    g1[7] = 0;                                           // stride1[47:16]
    i32x4 z4 = {};
    i32x8 z8 = {};
    __builtin_amdgcn_tensor_load_to_lds(g0, g1, z4, z4, z8, 0);
}

__global__ __launch_bounds__(256, 1)
void flow_kernel(const float* __restrict__ x,
                 const unsigned char* __restrict__ wsrc,
                 float* __restrict__ out) {
    extern __shared__ char smem[];
    _Float16* act1 = (_Float16*)(smem + LDS_ACT1);                 // [128][264] f16 (u1/v1)
    _Float16* act2 = (_Float16*)(smem + LDS_ACT2);                 // [128][264] f16 (u2/v2)
    _Float16* w1t  = (_Float16*)(smem + LDS_WBLOB);                // [128][264] f16
    _Float16* w2t  = (_Float16*)(smem + LDS_WBLOB + W1T_BYTES);    // [256][136] f16
    float*    b1s  = (float*)(smem + LDS_WBLOB + W1T_BYTES + W2T_BYTES);
    float*    b2s  = b1s + 128;
    _Float16* tstg = (_Float16*)(smem + LDS_TSTG);                 // [8][16][72] f16

    const int tid  = threadIdx.x;
    const int lane = tid & 31;
    const int wv   = tid >> 5;
    const int lr   = lane & 15;
    const int lh   = lane >> 4;
    const long row0 = (long)blockIdx.x * WROWS;

    // ---------------- load x tile (coalesced b32), f32 -> f16 LDS ----------------
    for (int r = 0; r < 16; ++r) {
        const int row = wv * 16 + r;
        const float* xp = x + (row0 + row) * XSTR;
        #pragma unroll
        for (int i = 0; i < 8; ++i)
            act1[row * AST + i * 32 + lane] = (_Float16)xp[i * 32 + lane];
        #pragma unroll
        for (int i = 0; i < 8; ++i)
            act2[row * AST + i * 32 + lane] = (_Float16)xp[256 + i * 32 + lane];
    }

    for (int j = 0; j < L_LAYERS; ++j) {
        __syncthreads();   // all waves done with previous layer's weights
        if (wv == 0) {     // wave-uniform: one TDM descriptor per layer
            tdm_load_blob(wsrc + (size_t)j * BLOB_BYTES,
                          (unsigned)(uintptr_t)(smem + LDS_WBLOB));
            __builtin_amdgcn_s_wait_tensorcnt((short)0);
        }
        __syncthreads();

        // ---------------- two coupling phases ----------------
        #pragma unroll 1
        for (int ph = 0; ph < 2; ++ph) {
            const _Float16* aIn = ph ? act1 : act2;   // MLP input (u2, then v1)
            _Float16*       aU  = ph ? act2 : act1;   // half updated elementwise
            const int ms = ph ? 0 : 1;                // scale-MLP index
            const int mt = ph ? 2 : 3;                // shift-MLP index
            const int mbase = ms;
            const int rbase = wv * 16;

            // A fragments (16x256 f16; A-layout K = kc*32 + lh*16 + e)
            v16h afr[8];
            {
                const _Float16* rp = aIn + (rbase + lr) * AST;
                #pragma unroll
                for (int kc = 0; kc < 8; ++kc) {
                    union { v16h v; v8h h[2]; } u;
                    u.h[0] = *(const v8h*)(rp + kc * 32 + lh * 16);
                    u.h[1] = *(const v8h*)(rp + kc * 32 + lh * 16 + 8);
                    afr[kc] = u.v;
                }
            }

            // GEMM1: [16x256] @ [256x64] -> 16x64 (cols 0..31 scale, 32..63 shift)
            v8f acc1[4];
            #pragma unroll
            for (int nt = 0; nt < 4; ++nt) {
                const int n   = nt * 16 + lr;
                const int mlp = mbase + 2 * (n >> 5);
                const _Float16* bp = w1t + (mlp * 32 + (n & 31)) * W1ST + lh * 16;
                v8f a = {};
                #pragma unroll
                for (int kc = 0; kc < 8; ++kc) {
                    union { v16h v; v8h h[2]; } bfr;
                    bfr.h[0] = *(const v8h*)(bp + kc * 32);
                    bfr.h[1] = *(const v8h*)(bp + kc * 32 + 8);
                    a = __builtin_amdgcn_wmma_f32_16x16x32_f16(false, afr[kc], false, bfr.v,
                                                               (short)0, a, false, false);
                }
                acc1[nt] = a;
            }

            // bias + tanh -> staging (C->A layout bounce)
            _Float16* st = tstg + wv * 16 * TST;
            #pragma unroll
            for (int nt = 0; nt < 4; ++nt) {
                const int n   = nt * 16 + lr;
                const int mlp = mbase + 2 * (n >> 5);
                const float bb = b1s[mlp * 32 + (n & 31)];
                #pragma unroll
                for (int v = 0; v < 8; ++v) {
                    const int row = v + 8 * lh;
                    st[row * TST + n] = (_Float16)fast_tanh(acc1[nt][v] + bb);
                }
            }

            // GEMM2 A fragments (16x32, K = H)
            union { v16h v; v8h h[2]; } a2s, a2t;
            {
                const _Float16* sp = st + lr * TST;
                a2s.h[0] = *(const v8h*)(sp + lh * 16);
                a2s.h[1] = *(const v8h*)(sp + lh * 16 + 8);
                a2t.h[0] = *(const v8h*)(sp + 32 + lh * 16);
                a2t.h[1] = *(const v8h*)(sp + 32 + lh * 16 + 8);
            }

            // GEMM2 (one WMMA per 16-col tile) + fused coupling update
            #pragma unroll 2
            for (int nt = 0; nt < 16; ++nt) {
                const int s = nt * 16 + lr;
                union { v16h v; v8h h[2]; } bS, bT;
                const _Float16* ps = w2t + s * W2ST + ms * 32 + lh * 16;
                const _Float16* pt = w2t + s * W2ST + mt * 32 + lh * 16;
                bS.h[0] = *(const v8h*)(ps);  bS.h[1] = *(const v8h*)(ps + 8);
                bT.h[0] = *(const v8h*)(pt);  bT.h[1] = *(const v8h*)(pt + 8);
                v8f z = {};
                v8f aS = __builtin_amdgcn_wmma_f32_16x16x32_f16(false, a2s.v, false, bS.v,
                                                                (short)0, z, false, false);
                v8f aT = __builtin_amdgcn_wmma_f32_16x16x32_f16(false, a2t.v, false, bT.v,
                                                                (short)0, z, false, false);
                const float cs = b2s[ms * 256 + s];
                const float ct = b2s[mt * 256 + s];
                _Float16* up = aU + rbase * AST + s;
                #pragma unroll
                for (int v = 0; v < 8; ++v) {
                    const int row = v + 8 * lh;
                    const float uv = (float)up[row * AST];
                    const float nv = uv * fast_exp(aS[v] + cs) + (aT[v] + ct);
                    up[row * AST] = (_Float16)nv;
                }
            }
        }
    }

    // ---------------- epilogue: out = [v1 | v2 | t] ----------------
    for (int r = 0; r < 16; ++r) {
        const int row = wv * 16 + r;
        float* op = out + (row0 + row) * XSTR;
        #pragma unroll
        for (int i = 0; i < 8; ++i)
            op[i * 32 + lane] = (float)act1[row * AST + i * 32 + lane];
        #pragma unroll
        for (int i = 0; i < 8; ++i)
            op[256 + i * 32 + lane] = (float)act2[row * AST + i * 32 + lane];
    }
    if (tid < WROWS) {  // t passthrough, one pass per block
        const long row = row0 + tid;
        out[row * XSTR + 512] = x[row * XSTR + 512];
    }
}

extern "C" void kernel_launch(void* const* d_in, const int* in_sizes, int n_in,
                              void* d_out, int out_size, void* d_ws, size_t ws_size,
                              hipStream_t stream) {
    const float* x  = (const float*)d_in[0];
    const float* W1 = (const float*)d_in[1];
    const float* b1 = (const float*)d_in[2];
    const float* W2 = (const float*)d_in[3];
    const float* b2 = (const float*)d_in[4];
    float* o = (float*)d_out;
    unsigned char* ws = (unsigned char*)d_ws;   // needs 3 * 141824 = 425,472 B

    prep_weights<<<dim3(L_LAYERS), dim3(256), 0, stream>>>(W1, b1, W2, b2, ws);
    flow_kernel<<<dim3(131072 / WROWS), dim3(256), LDS_TOTAL, stream>>>(x, ws, o);
}
